// BinaryBasicBlock_71141838291231
// MI455X (gfx1250) — 8-bit matrix operand order repaired and checked by reading, compiled, not run on hardware
//
#include <hip/hip_runtime.h>
#include <stdint.h>

typedef __attribute__((ext_vector_type(8))) int  v8i;
typedef __attribute__((ext_vector_type(4))) int  v4i;
typedef __attribute__((ext_vector_type(2))) int  v2i;

#define NB   32
#define CH   256
#define HW   56
#define PIX  (HW*HW)
#define NPIX (NB*PIX)
#define NELEM ((size_t)NB*CH*PIX)

static __device__ __forceinline__ float blockReduceSum(float v, float* red) {
  const int t = threadIdx.x;
  red[t] = v;
  __syncthreads();
  for (int s = 128; s > 0; s >>= 1) {
    if (t < s) red[t] += red[t + s];
    __syncthreads();
  }
  return red[0];
}

__global__ __launch_bounds__(256) void k_mean_nc(const float* __restrict__ x,
                                                 float* __restrict__ mean1) {
  __shared__ float red[256];
  const int b = blockIdx.x;
  const size_t base = (size_t)b * PIX;
  float s = 0.f;
  for (int i = threadIdx.x; i < PIX; i += 256) s += x[base + i];
  s = blockReduceSum(s, red);
  if (threadIdx.x == 0) mean1[b] = s * (1.f / PIX);
}

__global__ __launch_bounds__(256) void k_binarize1(const float* __restrict__ x,
                                                   const float* __restrict__ mean1,
                                                   signed char* __restrict__ xb,
                                                   float* __restrict__ beta_acc) {
  __shared__ float red[256];
  const size_t idx = (size_t)blockIdx.x * 256 + threadIdx.x;
  const int n  = (int)(idx / ((size_t)CH * PIX));
  const int c  = (int)((idx / PIX) % CH);
  const int hw = (int)(idx % PIX);
  const float s = x[idx] - mean1[n * CH + c];
  xb[((size_t)(n * PIX + hw)) * CH + c] = s > 0.f ? 1 : (s < 0.f ? -1 : 0);
  const float a = blockReduceSum(fabsf(s), red);
  if (threadIdx.x == 0) atomicAdd(beta_acc, a);
}

__global__ void k_beta_fin(const float* __restrict__ acc, float* __restrict__ beta) {
  *beta = *acc * (1.f / (float)NELEM);
}

__global__ __launch_bounds__(256) void k_pack_w(const float* __restrict__ w,
                                                signed char* __restrict__ wbf) {
  const int t = blockIdx.x * 256 + threadIdx.x;
  if (t >= 16 * 9 * 4 * 32) return;
  const int lane  = t & 31;
  const int kc    = (t >> 5) & 3;
  const int tap   = (t >> 7) % 9;
  const int ntile = t / 1152;
  const int hi    = lane >> 4;
  const int ncol  = lane & 15;
  const int co    = ntile * 16 + ncol;
  int words[8];
  for (int v = 0; v < 8; ++v) {
    unsigned wrd = 0;
    for (int j = 0; j < 4; ++j) {
      const int K  = (v < 4) ? (hi * 16 + v * 4 + j) : (32 + hi * 16 + (v - 4) * 4 + j);
      const int ci = kc * 64 + K;
      const float wv = w[((size_t)(co * CH + ci)) * 9 + tap];
      const signed char sgn = wv > 0.f ? 1 : (wv < 0.f ? -1 : 0);
      wrd |= ((unsigned)(unsigned char)sgn) << (8 * j);
    }
    words[v] = (int)wrd;
  }
  int* dst = (int*)(wbf + (((size_t)(ntile * 9 + tap) * 4 + kc) * 32 + lane) * 32);
  for (int v = 0; v < 8; ++v) dst[v] = words[v];
}

__global__ __launch_bounds__(512) void k_bconv(const signed char* __restrict__ xb,
                                               const signed char* __restrict__ wbf,
                                               const float* __restrict__ alpha,
                                               const float* __restrict__ beta,
                                               float* __restrict__ y) {
  __shared__ __align__(16) signed char sA[3 * 18 * CH];
  const int t   = threadIdx.x;
  const int tid = blockIdx.x;
  const int n   = tid / (HW * 4);
  const int rem = tid - n * (HW * 4);
  const int h   = rem >> 2;
  const int w0  = (rem & 3) << 4;

  for (int i = t; i < 54 * 16; i += 512) {
    const int pix = i >> 4;
    const int off = (i & 15) << 4;
    const int r   = pix / 18;
    const int wl  = pix - r * 18;
    const int hh  = h - 1 + r;
    const int ww  = w0 - 1 + wl;
    if (hh >= 0 && hh < HW && ww >= 0 && ww < HW) {
      const uint64_t gaddr =
          (uint64_t)(xb + ((size_t)((n * HW + hh) * HW + ww)) * CH + off);
      const unsigned ldsOff = (unsigned)(size_t)(sA + pix * CH + off);
      asm volatile("global_load_async_to_lds_b128 %0, %1, off"
                   :: "v"(ldsOff), "v"(gaddr) : "memory");
    } else {
      v4i z = {0, 0, 0, 0};
      *(v4i*)(sA + pix * CH + off) = z;
    }
  }
  asm volatile("s_wait_asynccnt 0" ::: "memory");
  __syncthreads();

  const int lane  = t & 31;
  const int ntile = t >> 5;
  const int lm    = lane & 15;
  const int hi    = lane >> 4;

  v8i acc = {0, 0, 0, 0, 0, 0, 0, 0};
  for (int tap = 0; tap < 9; ++tap) {
    const int ky = tap / 3;
    const int kx = tap - ky * 3;
    const int rowbase = (ky * 18 + lm + kx) * CH;
    const signed char* bptr =
        wbf + (((size_t)ntile * 9 + tap) * 4 * 32 + lane) * 32;
    for (int kc = 0; kc < 4; ++kc) {
      const int ab = rowbase + (kc << 6) + (hi << 4);
      v2i q0 = *(const v2i*)(sA + ab);
      v2i q1 = *(const v2i*)(sA + ab + 8);
      v2i q2 = *(const v2i*)(sA + ab + 32);
      v2i q3 = *(const v2i*)(sA + ab + 40);
      v8i a;
      a[0] = q0.x; a[1] = q0.y; a[2] = q1.x; a[3] = q1.y;
      a[4] = q2.x; a[5] = q2.y; a[6] = q3.x; a[7] = q3.y;
      const v4i* bp = (const v4i*)(bptr + (size_t)kc * 32 * 32);
      v4i b0 = bp[0];
      v4i b1 = bp[1];
      v8i b;
      b[0] = b0.x; b[1] = b0.y; b[2] = b0.z; b[3] = b0.w;
      b[4] = b1.x; b[5] = b1.y; b[6] = b1.z; b[7] = b1.w;
      acc = __builtin_amdgcn_wmma_i32_16x16x64_iu8(true, a, true, b, acc, false, false);
    }
  }

  const float sc = alpha[0] * beta[0];
  const int co = ntile * 16 + lm;
  for (int r = 0; r < 8; ++r) {
    const int mrow = hi * 8 + r;
    const int w = w0 + mrow;
    if (w < HW)
      y[((size_t)((n * HW + h) * HW + w)) * CH + co] = (float)acc[r] * sc;
  }
}

__global__ __launch_bounds__(256) void k_chan_stats(const float* __restrict__ y,
                                                    float* __restrict__ csum,
                                                    float* __restrict__ csq) {
  const int c  = threadIdx.x;
  const int p0 = blockIdx.x * (NPIX / 256);
  float s = 0.f, q = 0.f;
  for (int p = p0; p < p0 + NPIX / 256; ++p) {
    const float v = y[(size_t)p * CH + c];
    s += v; q += v * v;
  }
  atomicAdd(&csum[c], s);
  atomicAdd(&csq[c], q);
}

__global__ void k_bn_fin(const float* __restrict__ csum, const float* __restrict__ csq,
                         const float* __restrict__ gamma, const float* __restrict__ bb,
                         float* __restrict__ scale, float* __restrict__ shift) {
  const int c = threadIdx.x;
  const float inv = 1.f / (float)NPIX;
  const float m   = csum[c] * inv;
  const float var = csq[c] * inv - m * m;
  const float sc  = gamma[c] * rsqrtf(var + 1e-5f);
  scale[c] = sc;
  shift[c] = bb[c] - m * sc;
}

__global__ __launch_bounds__(256) void k_bnrelu_mean(const float* __restrict__ y,
                                                     const float* __restrict__ scale,
                                                     const float* __restrict__ shift,
                                                     float* __restrict__ mean2) {
  __shared__ float red[256];
  const int b = blockIdx.x;
  const int n = b >> 8;
  const int c = b & 255;
  const float sc = scale[c], sh = shift[c];
  float s = 0.f;
  for (int i = threadIdx.x; i < PIX; i += 256)
    s += fmaxf(y[((size_t)(n * PIX + i)) * CH + c] * sc + sh, 0.f);
  s = blockReduceSum(s, red);
  if (threadIdx.x == 0) mean2[b] = s * (1.f / PIX);
}

__global__ __launch_bounds__(256) void k_binarize2(const float* __restrict__ y,
                                                   const float* __restrict__ scale,
                                                   const float* __restrict__ shift,
                                                   const float* __restrict__ mean2,
                                                   signed char* __restrict__ xb,
                                                   float* __restrict__ beta_acc) {
  __shared__ float red[256];
  const size_t idx = (size_t)blockIdx.x * 256 + threadIdx.x;
  const int c = (int)(idx & 255);
  const int p = (int)(idx >> 8);
  const int n = p / PIX;
  const float z = fmaxf(y[idx] * scale[c] + shift[c], 0.f);
  const float s = z - mean2[n * CH + c];
  xb[idx] = s > 0.f ? 1 : (s < 0.f ? -1 : 0);
  const float a = blockReduceSum(fabsf(s), red);
  if (threadIdx.x == 0) atomicAdd(beta_acc, a);
}

__global__ __launch_bounds__(256) void k_final(const float* __restrict__ y,
                                               const float* __restrict__ scale,
                                               const float* __restrict__ shift,
                                               const float* __restrict__ x,
                                               float* __restrict__ out) {
  const size_t idx = (size_t)blockIdx.x * 256 + threadIdx.x;
  const int n  = (int)(idx / ((size_t)CH * PIX));
  const int c  = (int)((idx / PIX) % CH);
  const int hw = (int)(idx % PIX);
  const float v = y[((size_t)(n * PIX + hw)) * CH + c] * scale[c] + shift[c] + x[idx];
  out[idx] = fmaxf(v, 0.f);
}

extern "C" void kernel_launch(void* const* d_in, const int* in_sizes, int n_in,
                              void* d_out, int out_size, void* d_ws, size_t ws_size,
                              hipStream_t stream) {
  (void)in_sizes; (void)n_in; (void)out_size; (void)ws_size;

  const float* x        = (const float*)d_in[0];
  const float* conv1_w  = (const float*)d_in[1];
  const float* alpha1   = (const float*)d_in[2];
  const float* bn1_g    = (const float*)d_in[3];
  const float* bn1_b    = (const float*)d_in[4];
  const float* conv2_w  = (const float*)d_in[5];
  const float* alpha2   = (const float*)d_in[6];
  const float* bn2_g    = (const float*)d_in[7];
  const float* bn2_b    = (const float*)d_in[8];
  float* out            = (float*)d_out;

  char* ws = (char*)d_ws;
  size_t off = 0;
  auto alloc = [&](size_t bytes) { size_t o = off; off += (bytes + 255) & ~(size_t)255; return o; };

  float*       mean1     = (float*)(ws + alloc(NB * CH * 4));
  float*       mean2     = (float*)(ws + alloc(NB * CH * 4));
  float*       scal      = (float*)(ws + alloc(256));
  float*       ch_sum    = (float*)(ws + alloc(CH * 4));
  float*       ch_sq     = (float*)(ws + alloc(CH * 4));
  float*       bn1_scale = (float*)(ws + alloc(CH * 4));
  float*       bn1_shift = (float*)(ws + alloc(CH * 4));
  float*       bn2_scale = (float*)(ws + alloc(CH * 4));
  float*       bn2_shift = (float*)(ws + alloc(CH * 4));
  signed char* wbf1      = (signed char*)(ws + alloc(16 * 9 * 4 * 32 * 32));
  signed char* wbf2      = (signed char*)(ws + alloc(16 * 9 * 4 * 32 * 32));
  signed char* xb1       = (signed char*)(ws + alloc(NELEM));
  signed char* xb2       = (signed char*)(ws + alloc(NELEM));
  float*       y         = (float*)(ws + alloc(NELEM * 4));

  float* beta1_acc = scal + 0;
  float* beta1     = scal + 1;
  float* beta2_acc = scal + 2;
  float* beta2     = scal + 3;

  const int convBlocks  = NB * HW * 4;
  const int nElemBlocks = (int)(NELEM / 256);

  hipMemsetAsync(scal, 0, 16, stream);

  k_mean_nc  <<<NB * CH, 256, 0, stream>>>(x, mean1);
  k_binarize1<<<nElemBlocks, 256, 0, stream>>>(x, mean1, xb1, beta1_acc);
  k_beta_fin <<<1, 1, 0, stream>>>(beta1_acc, beta1);
  k_pack_w   <<<72, 256, 0, stream>>>(conv1_w, wbf1);
  k_pack_w   <<<72, 256, 0, stream>>>(conv2_w, wbf2);
  k_bconv    <<<convBlocks, 512, 0, stream>>>(xb1, wbf1, alpha1, beta1, y);

  hipMemsetAsync(ch_sum, 0, CH * 4, stream);
  hipMemsetAsync(ch_sq,  0, CH * 4, stream);
  k_chan_stats <<<256, 256, 0, stream>>>(y, ch_sum, ch_sq);
  k_bn_fin     <<<1, 256, 0, stream>>>(ch_sum, ch_sq, bn1_g, bn1_b, bn1_scale, bn1_shift);
  k_bnrelu_mean<<<NB * CH, 256, 0, stream>>>(y, bn1_scale, bn1_shift, mean2);
  k_binarize2  <<<nElemBlocks, 256, 0, stream>>>(y, bn1_scale, bn1_shift, mean2, xb2, beta2_acc);
  k_beta_fin   <<<1, 1, 0, stream>>>(beta2_acc, beta2);

  k_bconv<<<convBlocks, 512, 0, stream>>>(xb2, wbf2, alpha2, beta2, y);

  hipMemsetAsync(ch_sum, 0, CH * 4, stream);
  hipMemsetAsync(ch_sq,  0, CH * 4, stream);
  k_chan_stats<<<256, 256, 0, stream>>>(y, ch_sum, ch_sq);
  k_bn_fin    <<<1, 256, 0, stream>>>(ch_sum, ch_sq, bn2_g, bn2_b, bn2_scale, bn2_shift);
  k_final     <<<nElemBlocks, 256, 0, stream>>>(y, bn2_scale, bn2_shift, x, out);
}
